// BertAlibiLayer_69698729279728
// MI455X (gfx1250) — compile-verified, matrix operand order checked by reading, not run on hardware
//
#include <hip/hip_runtime.h>
#include <stdint.h>

// Problem constants (fixed by the reference)
#define B_    16
#define S_    512
#define DIM_  768
#define H_    12
#define HD_   64
#define I_    3072
#define MTOT  (B_ * S_)        // 8192 rows
#define QKVN  (3 * DIM_)       // 2304

typedef __bf16 bf16_t;
typedef __attribute__((ext_vector_type(16))) __bf16 v16bf;
typedef __attribute__((ext_vector_type(8)))  float  v8f;

__device__ __forceinline__ unsigned short f32_to_bf16(float f) {
  union { float f; uint32_t u; } c; c.f = f;
  uint32_t u = c.u;
  u += 0x7fffu + ((u >> 16) & 1u);       // round-to-nearest-even
  return (unsigned short)(u >> 16);
}

union FragB { uint4 q[2]; v16bf v; };    // 16 bf16 = one WMMA A/B fragment per lane

__device__ __forceinline__ v8f v8f_zero() {
  v8f z = {0.f, 0.f, 0.f, 0.f, 0.f, 0.f, 0.f, 0.f};
  return z;
}

// ---------------------------------------------------------------------------
// fp32 -> bf16 conversion (grid-stride)
// ---------------------------------------------------------------------------
__global__ void cvt_bf16_kernel(const float* __restrict__ in,
                                unsigned short* __restrict__ out, size_t n) {
  size_t i = (size_t)blockIdx.x * blockDim.x + threadIdx.x;
  size_t stride = (size_t)gridDim.x * blockDim.x;
  for (; i < n; i += stride) out[i] = f32_to_bf16(in[i]);
}

// ---------------------------------------------------------------------------
// Generic BF16 WMMA GEMM:  C[M,N] = A[M,K] * W[N,K]^T (+bias[N]) (+res[M,N])
// 128x128 tile / block (8 waves), K staged in LDS chunks of 64.
// Register double-buffered: next K-tile is fetched into VGPRs while the
// current tile's 16 WMMAs execute; LDS commit happens at loop top.
// ---------------------------------------------------------------------------
template <bool HAS_BIAS, bool HAS_RES, bool OUT_F32, bool OUT_BF16>
__global__ __launch_bounds__(256)
void gemm_bf16_kernel(const unsigned short* __restrict__ A,
                      const unsigned short* __restrict__ W,
                      const float* __restrict__ bias,
                      const float* __restrict__ res,
                      float* __restrict__ outF,
                      unsigned short* __restrict__ outB,
                      int M, int N, int K) {
  constexpr int LST = 72;                       // padded LDS stride (bf16 elems)
  __shared__ unsigned short sA[128 * LST];
  __shared__ unsigned short sB[128 * LST];

  const int tid  = threadIdx.x;
  const int wave = tid >> 5;
  const int lane = tid & 31;
  const int m0 = blockIdx.y * 128;
  const int n0 = blockIdx.x * 128;
  const int wm = (wave >> 1) * 32;              // 0,32,64,96
  const int wn = (wave & 1) * 64;               // 0,64
  const int l15 = lane & 15;
  const int hi8 = (lane >> 4) << 3;             // 0 or 8

  v8f acc[2][4];
  for (int mi = 0; mi < 2; ++mi)
    for (int ni = 0; ni < 4; ++ni) acc[mi][ni] = v8f_zero();

  const int lr = tid >> 1;                      // staging row 0..127
  const int lc = (tid & 1) * 32;                // staging col 0 / 32
  const unsigned short* gA = A + (size_t)(m0 + lr) * K + lc;
  const unsigned short* gW = W + (size_t)(n0 + lr) * K + lc;

  uint4 ra[4], rb[4];                           // register staging buffers
  {
    ra[0] = *(const uint4*)(gA);      ra[1] = *(const uint4*)(gA + 8);
    ra[2] = *(const uint4*)(gA + 16); ra[3] = *(const uint4*)(gA + 24);
    rb[0] = *(const uint4*)(gW);      rb[1] = *(const uint4*)(gW + 8);
    rb[2] = *(const uint4*)(gW + 16); rb[3] = *(const uint4*)(gW + 24);
  }

  for (int k0 = 0; k0 < K; k0 += 64) {
    __syncthreads();                            // prev compute done reading LDS
    {
      unsigned short* la = &sA[lr * LST + lc];
      unsigned short* lb = &sB[lr * LST + lc];
      *(uint4*)(la)      = ra[0]; *(uint4*)(la + 8)  = ra[1];
      *(uint4*)(la + 16) = ra[2]; *(uint4*)(la + 24) = ra[3];
      *(uint4*)(lb)      = rb[0]; *(uint4*)(lb + 8)  = rb[1];
      *(uint4*)(lb + 16) = rb[2]; *(uint4*)(lb + 24) = rb[3];
    }
    __syncthreads();                            // tile visible to all waves

    if (k0 + 64 < K) {                          // prefetch next tile into regs
      const unsigned short* na = gA + k0 + 64;
      const unsigned short* nw = gW + k0 + 64;
      ra[0] = *(const uint4*)(na);      ra[1] = *(const uint4*)(na + 8);
      ra[2] = *(const uint4*)(na + 16); ra[3] = *(const uint4*)(na + 24);
      rb[0] = *(const uint4*)(nw);      rb[1] = *(const uint4*)(nw + 8);
      rb[2] = *(const uint4*)(nw + 16); rb[3] = *(const uint4*)(nw + 24);
      if (k0 + 128 < K) {                       // keep L2 warm two tiles ahead
        __builtin_prefetch(na + 64, 0, 1);
        __builtin_prefetch(nw + 64, 0, 1);
      }
    }

    for (int ks = 0; ks < 64; ks += 32) {
      const int c0 = ks + hi8;
      FragB a[2], b[4];
      for (int mi = 0; mi < 2; ++mi) {
        const unsigned short* p = &sA[(wm + mi * 16 + l15) * LST + c0];
        a[mi].q[0] = *(const uint4*)p;
        a[mi].q[1] = *(const uint4*)(p + 16);
      }
      for (int ni = 0; ni < 4; ++ni) {
        const unsigned short* p = &sB[(wn + ni * 16 + l15) * LST + c0];
        b[ni].q[0] = *(const uint4*)p;
        b[ni].q[1] = *(const uint4*)(p + 16);
      }
      for (int mi = 0; mi < 2; ++mi)
        for (int ni = 0; ni < 4; ++ni)
          acc[mi][ni] = __builtin_amdgcn_wmma_f32_16x16x32_bf16(
              false, a[mi].v, false, b[ni].v, (short)0, acc[mi][ni], false, false);
    }
  }

  // Epilogue: C layout: vgpr j -> M=j (lanes 0-15) / M=j+8 (lanes 16-31); N=lane&15
  const int rowb = m0 + wm + hi8;
  const int colb = n0 + wn + l15;
  for (int mi = 0; mi < 2; ++mi) {
    for (int ni = 0; ni < 4; ++ni) {
      const int col = colb + ni * 16;
      float bv = HAS_BIAS ? bias[col] : 0.f;
      for (int j = 0; j < 8; ++j) {
        const int row = rowb + mi * 16 + j;
        float v = acc[mi][ni][j] + bv;
        if (HAS_RES) v += res[(size_t)row * N + col];
        if (OUT_F32) outF[(size_t)row * N + col] = v;
        if (OUT_BF16) outB[(size_t)row * N + col] = f32_to_bf16(v);
      }
    }
  }
}

// ---------------------------------------------------------------------------
// Fused attention: per-(b,h,64-query-block), everything in LDS.
//   QK^T (wmma) -> +alibi, *1/8 -> softmax (in LDS) -> P*V (wmma) -> ctx bf16
// Dynamic LDS map (bytes):
//   sQ  : 64*72  bf16 =   9216
//   sK  : 128*72 bf16 =  18432
//   sS  : 64*512 f32  = 131072
//   sP  : 64*520 bf16 =  66560
//   sVt : 64*520 bf16 =  66560   -> total 291840 (<320KB/WG)
// ---------------------------------------------------------------------------
__global__ __launch_bounds__(256)
void attn_kernel(const unsigned short* __restrict__ qkv,   // [MTOT, 3*DIM]
                 const float* __restrict__ slopes,         // [H]
                 unsigned short* __restrict__ ctx)         // [MTOT, DIM] bf16
{
  extern __shared__ unsigned char smem_raw[];
  unsigned short* sQ  = (unsigned short*)smem_raw;            // 64*72
  unsigned short* sK  = sQ + 64 * 72;                         // 128*72
  float*          sS  = (float*)(sK + 128 * 72);              // 64*512
  unsigned short* sP  = (unsigned short*)(sS + 64 * 512);     // 64*520
  unsigned short* sVt = sP + 64 * 520;                        // 64*520 (V^T)

  const int tid  = threadIdx.x;
  const int wave = tid >> 5;
  const int lane = tid & 31;
  const int l15  = lane & 15;
  const int hi8  = (lane >> 4) << 3;
  const int qb = blockIdx.x;      // 0..7   (64-row query block)
  const int h  = blockIdx.y;      // 0..11
  const int b  = blockIdx.z;      // 0..15
  const float slope = slopes[h];

  const size_t rstride = QKVN;
  const unsigned short* Qg = qkv + (size_t)(b * S_ + qb * 64) * rstride + h * HD_;
  const unsigned short* Kg = qkv + (size_t)(b * S_) * rstride + DIM_ + h * HD_;
  const unsigned short* Vg = qkv + (size_t)(b * S_) * rstride + 2 * DIM_ + h * HD_;

  // Stage Q (64x64): thread -> row tid>>2, 16-col chunk (tid&3)*16
  {
    int r = tid >> 2, c = (tid & 3) * 16;
    const unsigned short* g = Qg + (size_t)r * rstride + c;
    uint4 d0 = *(const uint4*)g;
    uint4 d1 = *(const uint4*)(g + 8);
    *(uint4*)&sQ[r * 72 + c]     = d0;
    *(uint4*)&sQ[r * 72 + c + 8] = d1;
  }
  // Stage V transposed: 512t x 64d -> sVt[d][t]
  for (int it = 0; it < 8; ++it) {
    int c = tid + it * 256;                 // 0..2047 tasks
    int t = c >> 2, dc = (c & 3) * 16;
    const unsigned short* g = Vg + (size_t)t * rstride + dc;
    union { uint4 q; unsigned short s[8]; } u0, u1;
    u0.q = *(const uint4*)g;
    u1.q = *(const uint4*)(g + 8);
    for (int i = 0; i < 8; ++i) sVt[(dc + i) * 520 + t]     = u0.s[i];
    for (int i = 0; i < 8; ++i) sVt[(dc + 8 + i) * 520 + t] = u1.s[i];
  }

  // ---- Scores: S[64x512] = Q K^T / 8 - slope*|s-t| ----
  const int wm = (wave >> 1) * 16;   // 0..48 : wave's 16 query rows
  const int wn = (wave & 1) * 64;    // 0/64  : wave's 64-key strip
  for (int kt = 0; kt < 4; ++kt) {
    __syncthreads();
    for (int it = 0; it < 2; ++it) {                 // stage 128x64 K tile
      int c = tid + it * 256;                        // 512 tasks
      int r = c >> 2, dc = (c & 3) * 16;
      const unsigned short* g = Kg + (size_t)(kt * 128 + r) * rstride + dc;
      uint4 d0 = *(const uint4*)g;
      uint4 d1 = *(const uint4*)(g + 8);
      *(uint4*)&sK[r * 72 + dc]     = d0;
      *(uint4*)&sK[r * 72 + dc + 8] = d1;
    }
    __syncthreads();

    v8f acc[4];
    for (int ni = 0; ni < 4; ++ni) acc[ni] = v8f_zero();
    for (int ks = 0; ks < 64; ks += 32) {
      const int c0 = ks + hi8;
      FragB a, bf[4];
      const unsigned short* pa = &sQ[(wm + l15) * 72 + c0];
      a.q[0] = *(const uint4*)pa;
      a.q[1] = *(const uint4*)(pa + 16);
      for (int ni = 0; ni < 4; ++ni) {
        const unsigned short* pb = &sK[(wn + ni * 16 + l15) * 72 + c0];
        bf[ni].q[0] = *(const uint4*)pb;
        bf[ni].q[1] = *(const uint4*)(pb + 16);
      }
      for (int ni = 0; ni < 4; ++ni)
        acc[ni] = __builtin_amdgcn_wmma_f32_16x16x32_bf16(
            false, a.v, false, bf[ni].v, (short)0, acc[ni], false, false);
    }
    // write to LDS with alibi + scale
    for (int ni = 0; ni < 4; ++ni) {
      const int tcol = kt * 128 + wn + ni * 16 + l15;     // local == global t
      for (int j = 0; j < 8; ++j) {
        const int srow = wm + hi8 + j;                    // local query row
        const int sglb = qb * 64 + srow;
        float v = acc[ni][j] * 0.125f - slope * fabsf((float)(sglb - tcol));
        sS[srow * 512 + tcol] = v;
      }
    }
  }
  __syncthreads();

  // ---- Softmax over each 512-row, wave-per-row (8 rows / wave) ----
  for (int rr = 0; rr < 8; ++rr) {
    const int r = wave * 8 + rr;
    float vals[16];
    float mx = -1e30f;
    for (int i = 0; i < 16; ++i) {
      vals[i] = sS[r * 512 + lane + i * 32];
      mx = fmaxf(mx, vals[i]);
    }
    for (int o = 16; o > 0; o >>= 1) mx = fmaxf(mx, __shfl_xor(mx, o, 32));
    float sum = 0.f;
    for (int i = 0; i < 16; ++i) { vals[i] = __expf(vals[i] - mx); sum += vals[i]; }
    for (int o = 16; o > 0; o >>= 1) sum += __shfl_xor(sum, o, 32);
    const float inv = 1.0f / sum;
    for (int i = 0; i < 16; ++i)
      sP[r * 520 + lane + i * 32] = f32_to_bf16(vals[i] * inv);
  }
  __syncthreads();

  // ---- ctx[64x64] = P[64x512] * V[512x64]  (B = V^T rows, contiguous K) ----
  const int pn = (wave & 1) * 32;    // wave's 32 output dims
  v8f acc2[2];
  acc2[0] = v8f_zero(); acc2[1] = v8f_zero();
  for (int ks = 0; ks < 512; ks += 32) {
    const int c0 = ks + hi8;
    FragB a, b0, b1;
    const unsigned short* pa = &sP[(wm + l15) * 520 + c0];
    a.q[0] = *(const uint4*)pa;
    a.q[1] = *(const uint4*)(pa + 16);
    const unsigned short* pb0 = &sVt[(pn + l15) * 520 + c0];
    b0.q[0] = *(const uint4*)pb0;
    b0.q[1] = *(const uint4*)(pb0 + 16);
    const unsigned short* pb1 = &sVt[(pn + 16 + l15) * 520 + c0];
    b1.q[0] = *(const uint4*)pb1;
    b1.q[1] = *(const uint4*)(pb1 + 16);
    acc2[0] = __builtin_amdgcn_wmma_f32_16x16x32_bf16(
        false, a.v, false, b0.v, (short)0, acc2[0], false, false);
    acc2[1] = __builtin_amdgcn_wmma_f32_16x16x32_bf16(
        false, a.v, false, b1.v, (short)0, acc2[1], false, false);
  }
  for (int ni = 0; ni < 2; ++ni) {
    const int d = pn + ni * 16 + l15;
    for (int j = 0; j < 8; ++j) {
      const int srow = qb * 64 + wm + hi8 + j;
      ctx[(size_t)(b * S_ + srow) * DIM_ + h * HD_ + d] = f32_to_bf16(acc2[ni][j]);
    }
  }
}

// ---------------------------------------------------------------------------
// LayerNorm over rows of 768; wave-per-row (24 elems/lane, wave32 shuffles)
// ---------------------------------------------------------------------------
template <bool OUT_F32, bool OUT_BF16>
__global__ __launch_bounds__(256)
void layernorm_kernel(const float* __restrict__ in,
                      const float* __restrict__ g,
                      const float* __restrict__ be,
                      float* __restrict__ outF,
                      unsigned short* __restrict__ outB, int M) {
  const int wave = threadIdx.x >> 5, lane = threadIdx.x & 31;
  const int row = blockIdx.x * 8 + wave;
  if (row >= M) return;
  const float* p = in + (size_t)row * DIM_;
  float v[24];
  float s = 0.f;
  for (int i = 0; i < 24; ++i) { v[i] = p[lane + i * 32]; s += v[i]; }
  for (int o = 16; o > 0; o >>= 1) s += __shfl_xor(s, o, 32);
  const float mean = s * (1.0f / 768.0f);
  float s2 = 0.f;
  for (int i = 0; i < 24; ++i) { float d = v[i] - mean; s2 += d * d; }
  for (int o = 16; o > 0; o >>= 1) s2 += __shfl_xor(s2, o, 32);
  const float inv = rsqrtf(s2 * (1.0f / 768.0f) + 1e-12f);
  for (int i = 0; i < 24; ++i) {
    const int c = lane + i * 32;
    float o = (v[i] - mean) * inv * g[c] + be[c];
    if (OUT_F32) outF[(size_t)row * DIM_ + c] = o;
    if (OUT_BF16) outB[(size_t)row * DIM_ + c] = f32_to_bf16(o);
  }
}

// ---------------------------------------------------------------------------
// GLU gate: gated = gelu_exact(glu[:, :I]) * glu[:, I:], out bf16
// ---------------------------------------------------------------------------
__global__ void glu_gate_kernel(const float* __restrict__ glu,
                                unsigned short* __restrict__ gated) {
  size_t i = (size_t)blockIdx.x * blockDim.x + threadIdx.x;
  const size_t total = (size_t)MTOT * I_;
  const size_t stride = (size_t)gridDim.x * blockDim.x;
  for (; i < total; i += stride) {
    const size_t m = i / I_, j = i % I_;
    const float gv = glu[m * (2 * I_) + j];
    const float uv = glu[m * (2 * I_) + I_ + j];
    const float ge = 0.5f * gv * (1.0f + erff(gv * 0.70710678118654752f));
    gated[i] = f32_to_bf16(ge * uv);
  }
}

// ---------------------------------------------------------------------------
// Orchestration
// ---------------------------------------------------------------------------
extern "C" void kernel_launch(void* const* d_in, const int* in_sizes, int n_in,
                              void* d_out, int out_size, void* d_ws, size_t ws_size,
                              hipStream_t stream) {
  const float* x      = (const float*)d_in[0];   // hidden_states [8192,768]
  const float* slopes = (const float*)d_in[4];
  const float* Wqkv   = (const float*)d_in[5];
  const float* bqkv   = (const float*)d_in[6];
  const float* Wo     = (const float*)d_in[7];
  const float* bo     = (const float*)d_in[8];
  const float* ln1g   = (const float*)d_in[9];
  const float* ln1b   = (const float*)d_in[10];
  const float* Wglu   = (const float*)d_in[11];
  const float* Wdown  = (const float*)d_in[12];
  const float* bdown  = (const float*)d_in[13];
  const float* ln2g   = (const float*)d_in[14];
  const float* ln2b   = (const float*)d_in[15];
  float* out = (float*)d_out;

  // workspace carve (256B aligned)
  uint8_t* base = (uint8_t*)d_ws;
  size_t off = 0;
  auto carve = [&](size_t bytes) -> uint8_t* {
    uint8_t* p = base + off;
    off = (off + bytes + 255) & ~(size_t)255;
    return p;
  };
  unsigned short* x_b     = (unsigned short*)carve((size_t)MTOT * DIM_ * 2);
  unsigned short* Wqkv_b  = (unsigned short*)carve((size_t)QKVN * DIM_ * 2);
  unsigned short* Wo_b    = (unsigned short*)carve((size_t)DIM_ * DIM_ * 2);
  unsigned short* Wglu_b  = (unsigned short*)carve((size_t)(2 * I_) * DIM_ * 2);
  unsigned short* Wdown_b = (unsigned short*)carve((size_t)DIM_ * I_ * 2);
  unsigned short* qkv_b   = (unsigned short*)carve((size_t)MTOT * QKVN * 2);
  unsigned short* ctx_b   = (unsigned short*)carve((size_t)MTOT * DIM_ * 2);
  float*          attnpre = (float*)carve((size_t)MTOT * DIM_ * 4);
  float*          attn_f  = (float*)carve((size_t)MTOT * DIM_ * 4);
  unsigned short* attn_b  = (unsigned short*)carve((size_t)MTOT * DIM_ * 2);
  float*          glu_f   = (float*)carve((size_t)MTOT * (2 * I_) * 4);
  unsigned short* gated_b = (unsigned short*)carve((size_t)MTOT * I_ * 2);
  float*          mlp_f   = (float*)carve((size_t)MTOT * DIM_ * 4);
  (void)ws_size; (void)in_sizes; (void)n_in; (void)out_size;

  // 1) fp32 -> bf16 conversions
  cvt_bf16_kernel<<<2048, 256, 0, stream>>>(x,     x_b,     (size_t)MTOT * DIM_);
  cvt_bf16_kernel<<<2048, 256, 0, stream>>>(Wqkv,  Wqkv_b,  (size_t)QKVN * DIM_);
  cvt_bf16_kernel<<<1024, 256, 0, stream>>>(Wo,    Wo_b,    (size_t)DIM_ * DIM_);
  cvt_bf16_kernel<<<2048, 256, 0, stream>>>(Wglu,  Wglu_b,  (size_t)(2 * I_) * DIM_);
  cvt_bf16_kernel<<<2048, 256, 0, stream>>>(Wdown, Wdown_b, (size_t)DIM_ * I_);

  // 2) QKV = x @ Wqkv^T + bqkv -> bf16 [8192, 2304]
  gemm_bf16_kernel<true, false, false, true>
      <<<dim3(QKVN / 128, MTOT / 128), 256, 0, stream>>>(
          x_b, Wqkv_b, bqkv, nullptr, nullptr, qkv_b, MTOT, QKVN, DIM_);

  // 3) fused attention (QK^T + alibi + softmax + PV), ctx bf16 [8192, 768]
  attn_kernel<<<dim3(S_ / 64, H_, B_), 256, 291840, stream>>>(qkv_b, slopes, ctx_b);

  // 4) attnpre = ctx @ Wo^T + bo + x  (fp32)
  gemm_bf16_kernel<true, true, true, false>
      <<<dim3(DIM_ / 128, MTOT / 128), 256, 0, stream>>>(
          ctx_b, Wo_b, bo, x, attnpre, nullptr, MTOT, DIM_, DIM_);

  // 5) attn_out = LN1(attnpre) -> fp32 + bf16
  layernorm_kernel<true, true><<<MTOT / 8, 256, 0, stream>>>(
      attnpre, ln1g, ln1b, attn_f, attn_b, MTOT);

  // 6) glu = attn_out @ Wglu^T  (fp32 [8192, 6144])
  gemm_bf16_kernel<false, false, true, false>
      <<<dim3((2 * I_) / 128, MTOT / 128), 256, 0, stream>>>(
          attn_b, Wglu_b, nullptr, nullptr, glu_f, nullptr, MTOT, 2 * I_, DIM_);

  // 7) gated = gelu(glu[:, :I]) * glu[:, I:] -> bf16
  glu_gate_kernel<<<8192, 256, 0, stream>>>(glu_f, gated_b);

  // 8) mlp = gated @ Wdown^T + bdown + attn_out  (fp32)
  gemm_bf16_kernel<true, true, true, false>
      <<<dim3(DIM_ / 128, MTOT / 128), 256, 0, stream>>>(
          gated_b, Wdown_b, bdown, attn_f, mlp_f, nullptr, MTOT, DIM_, I_);

  // 9) out = LN2(mlp) -> d_out fp32
  layernorm_kernel<true, false><<<MTOT / 8, 256, 0, stream>>>(
      mlp_f, ln2g, ln2b, out, nullptr, MTOT);
}
